// NodeHead_12438225290021
// MI455X (gfx1250) — compile-verified, matrix operand order checked by reading, not run on hardware
//
#include <hip/hip_runtime.h>
#include <hip/hip_bf16.h>
#include <math.h>

// ---------------------------------------------------------------------------
// Fused context-attention + pointer-logits head for MI455X (gfx1250).
//  - bf16 WMMA (v_wmma_f32_16x16x32_bf16) for the two large GEMMs (k,v proj)
//  - algebraic folding of the kp GEMM: logits[n] = X[n] . (qp @ W_k) / sqrt(E)
//  - online softmax per head (wave h == head h), wave32 shuffles
//  - global_prefetch_b8 for the next node tile
// ---------------------------------------------------------------------------

typedef __attribute__((ext_vector_type(16))) __bf16 v16bf;
typedef __attribute__((ext_vector_type(8)))  __bf16 bf16x8;
typedef __attribute__((ext_vector_type(8)))  float  v8f;

#define E    128
#define H    8
#define HD   16
#define NN   1000
#define BB   512
#define TILE 64
#define SX   136   // LDS row stride in bf16 elements (272B, 16B aligned)

__device__ __forceinline__ float wave_rmax(float v) {
#pragma unroll
  for (int m = 16; m >= 1; m >>= 1) v = fmaxf(v, __shfl_xor(v, m, 32));
  return v;
}
__device__ __forceinline__ float wave_rsum(float v) {
#pragma unroll
  for (int m = 16; m >= 1; m >>= 1) v += __shfl_xor(v, m, 32);
  return v;
}

// Build a 16x32 WMMA operand fragment (one lane's 16 bf16 values) from an
// fp32 row in global memory. CDNA5 16-bit operand layout: lanes 0-15 hold
// K={k0..k0+7, k0+16..k0+23}, lanes 16-31 hold K={k0+8..k0+15, k0+24..k0+31}.
__device__ __forceinline__ v16bf frag_from_f32row(const float* __restrict__ row,
                                                  int k0, int ho) {
  const float* p0 = row + k0 + ho;
  const float* p1 = row + k0 + ho + 16;
  v16bf f;
#pragma unroll
  for (int j = 0; j < 8; ++j) f[j] = (__bf16)p0[j];
#pragma unroll
  for (int j = 0; j < 8; ++j) f[8 + j] = (__bf16)p1[j];
  return f;
}

// Same fragment pattern loaded from a bf16 LDS row (two 16B chunks).
__device__ __forceinline__ v16bf frag_from_lds(const __bf16* rowp, int k0, int ho) {
  union { v16bf v; bf16x8 h[2]; } u;
  u.h[0] = *(const bf16x8*)(rowp + k0 + ho);
  u.h[1] = *(const bf16x8*)(rowp + k0 + ho + 16);
  return u.v;
}

// ---------------------------------------------------------------------------
// Kernel 1: q[b] = (ctx @ W_ctx^T) @ wq^T + bq        (tiny, VALU only)
// ---------------------------------------------------------------------------
__global__ __launch_bounds__(128)
void ctx_query_kernel(const float* __restrict__ graph_emb,
                      const float* __restrict__ soc,
                      const float* __restrict__ cap,
                      const float* __restrict__ tm,
                      const float* __restrict__ W_ctx,
                      const float* __restrict__ in_proj_w,
                      const float* __restrict__ in_proj_b,
                      float* __restrict__ qbuf) {
  __shared__ float ctx[E + 4];
  __shared__ float q0s[E];
  const int b = blockIdx.x, tid = threadIdx.x;
  ctx[tid] = graph_emb[(size_t)b * E + tid];
  if (tid == 0) {
    float s = soc[b], c = cap[b];
    ctx[E] = s; ctx[E + 1] = c; ctx[E + 2] = tm[b]; ctx[E + 3] = s * c;
  }
  __syncthreads();
  {
    float s = 0.f;
    const float* row = W_ctx + (size_t)tid * (E + 4);
#pragma unroll 4
    for (int c = 0; c < E + 4; ++c) s += ctx[c] * row[c];
    q0s[tid] = s;
  }
  __syncthreads();
  {
    float q = in_proj_b[tid];                       // bq
    const float* wrow = in_proj_w + (size_t)tid * E; // wq rows 0..127
#pragma unroll 4
    for (int c = 0; c < E; ++c) q += q0s[c] * wrow[c];
    qbuf[(size_t)b * E + tid] = q;
  }
}

// ---------------------------------------------------------------------------
// Kernel 2: per-batch fused attention + pointer logits.
// ---------------------------------------------------------------------------
__global__ __launch_bounds__(256)
void fused_attn_pointer_kernel(const float* __restrict__ node_emb,
                               const int*   __restrict__ mask,
                               const float* __restrict__ in_proj_w,
                               const float* __restrict__ in_proj_b,
                               const float* __restrict__ out_proj_w,
                               const float* __restrict__ out_proj_b,
                               const float* __restrict__ W_q,
                               const float* __restrict__ W_k,
                               const float* __restrict__ qbuf,
                               float*       __restrict__ out) {
  __shared__ __bf16 X_lds[TILE][SX];   // node tile (bf16)
  __shared__ __bf16 K_lds[TILE][SX];   // k tile
  __shared__ __bf16 V_lds[TILE][SX];   // v tile
  __shared__ float  q_lds[E];
  __shared__ float  p_lds[H][TILE];
  __shared__ float  att_lds[E];
  __shared__ float  att2_lds[E];
  __shared__ float  qp_lds[E];
  __shared__ float  r_lds[E];

  const int b    = blockIdx.x;
  const int tid  = threadIdx.x;
  const int wave = tid >> 5;           // wave == head == 16-col slab
  const int lane = tid & 31;
  const int l15  = lane & 15;
  const int ho   = (lane & 16) ? 8 : 0;

  const float* Xb = node_emb + (size_t)b * NN * E;
  const int*   Mb = mask + (size_t)b * NN;

  if (tid < E) q_lds[tid] = qbuf[(size_t)b * E + tid];

  // Persistent weight (B-matrix) fragments: column e of K/V output == row e
  // of wk/wv, same K-striping as the A operand. Kept in VGPRs all kernel.
  const int ecol = wave * 16 + l15;
  const float* wk_row = in_proj_w + (size_t)(E + ecol) * E;       // wk
  const float* wv_row = in_proj_w + (size_t)(2 * E + ecol) * E;   // wv
  v16bf bkF[4], bvF[4];
#pragma unroll
  for (int ks = 0; ks < 4; ++ks) {
    bkF[ks] = frag_from_f32row(wk_row, 32 * ks, ho);
    bvF[ks] = frag_from_f32row(wv_row, 32 * ks, ho);
  }
  const float bias_k = in_proj_b[E + ecol];
  const float bias_v = in_proj_b[2 * E + ecol];

  // Online-softmax running state for head == wave; acc holds dim d = lane&15.
  float m_run = -INFINITY, l_run = 0.0f, acc = 0.0f;

  const int NT   = (NN + TILE - 1) / TILE;   // 16 tiles (last partial)
  const int xrow = tid >> 2;                 // 0..63
  const int xq   = (tid & 3) * 32;           // column base within row

  for (int t = 0; t < NT; ++t) {
    const int n0 = t * TILE;

    // ---- stage X tile: fp32 global -> bf16 LDS (coalesced, vectorized) ----
    {
      int rg = n0 + xrow; if (rg > NN - 1) rg = NN - 1;   // clamp tail rows
      const float* src = Xb + (size_t)rg * E + xq;
#pragma unroll
      for (int s = 0; s < 4; ++s) {
        float4 f0 = *(const float4*)(src + 8 * s);
        float4 f1 = *(const float4*)(src + 8 * s + 4);
        bf16x8 pk;
        pk[0] = (__bf16)f0.x; pk[1] = (__bf16)f0.y;
        pk[2] = (__bf16)f0.z; pk[3] = (__bf16)f0.w;
        pk[4] = (__bf16)f1.x; pk[5] = (__bf16)f1.y;
        pk[6] = (__bf16)f1.z; pk[7] = (__bf16)f1.w;
        *(bf16x8*)&X_lds[xrow][xq + 8 * s] = pk;
      }
      if (t + 1 < NT) {                      // prefetch next tile into caches
        int rp = n0 + TILE + xrow; if (rp > NN - 1) rp = NN - 1;
        __builtin_prefetch(Xb + (size_t)rp * E + xq, 0, 1);
      }
    }
    __syncthreads();

    // ---- WMMA: K = X@wk^T, V = X@wv^T  (this wave: cols [16w,16w+16)) ----
    v8f accK[4], accV[4];
#pragma unroll
    for (int i = 0; i < 4; ++i)
#pragma unroll
      for (int r = 0; r < 8; ++r) { accK[i][r] = 0.f; accV[i][r] = 0.f; }

#pragma unroll
    for (int ks = 0; ks < 4; ++ks) {
#pragma unroll
      for (int i = 0; i < 4; ++i) {
        v16bf a = frag_from_lds(&X_lds[i * 16 + l15][0], 32 * ks, ho);
        accK[i] = __builtin_amdgcn_wmma_f32_16x16x32_bf16(
            false, a, false, bkF[ks], (short)0, accK[i], false, false);
        accV[i] = __builtin_amdgcn_wmma_f32_16x16x32_bf16(
            false, a, false, bvF[ks], (short)0, accV[i], false, false);
      }
    }
    // C/D layout: VGPR r -> row M = r (lanes 0-15) / r+8 (lanes 16-31).
#pragma unroll
    for (int i = 0; i < 4; ++i)
#pragma unroll
      for (int r = 0; r < 8; ++r) {
        int n = i * 16 + r + ho;
        K_lds[n][ecol] = (__bf16)(accK[i][r] + bias_k);
        V_lds[n][ecol] = (__bf16)(accV[i][r] + bias_v);
      }
    __syncthreads();

    // ---- scores + online softmax + p.V accumulation (head h == wave) ----
    {
      const int h = wave;
      float s0 = 0.f, s1 = 0.f;
#pragma unroll
      for (int d = 0; d < HD; ++d) {
        float qv = q_lds[h * HD + d];
        s0 += qv * (float)K_lds[lane][h * HD + d];
        s1 += qv * (float)K_lds[lane + 32][h * HD + d];
      }
      s0 *= 0.25f; s1 *= 0.25f;                    // 1/sqrt(HD)
      int gn0 = n0 + lane, gn1 = n0 + lane + 32;
      if (gn0 >= NN || Mb[gn0] != 0) s0 = -INFINITY;
      if (gn1 >= NN || Mb[gn1] != 0) s1 = -INFINITY;

      float tmax  = wave_rmax(fmaxf(s0, s1));
      float m_new = fmaxf(m_run, tmax);
      float scale = (m_run <= -1e37f) ? 0.f : __expf(m_run - m_new);
      float e0 = (s0 <= -1e37f) ? 0.f : __expf(s0 - m_new);
      float e1 = (s1 <= -1e37f) ? 0.f : __expf(s1 - m_new);
      l_run = l_run * scale + wave_rsum(e0 + e1);
      acc  *= scale;
      m_run = m_new;
      p_lds[h][lane]      = e0;                    // same-wave LDS producer
      p_lds[h][lane + 32] = e1;
#pragma unroll 4
      for (int n = 0; n < TILE; ++n)
        acc += p_lds[h][n] * (float)V_lds[n][h * HD + l15];
    }
    __syncthreads();
  }

  // attended = acc / l  (per head, dims 0..15 live in lanes 0..15)
  if (lane < 16) att_lds[wave * HD + lane] = acc / l_run;
  __syncthreads();

  // attended @ out_proj_w^T + b  -> att2
  if (tid < E) {
    float s = out_proj_b[tid];
    const float* wrow = out_proj_w + (size_t)tid * E;
#pragma unroll 4
    for (int c = 0; c < E; ++c) s += att_lds[c] * wrow[c];
    att2_lds[tid] = s;
  }
  __syncthreads();
  // qp = att2 @ W_q^T
  if (tid < E) {
    float s = 0.f;
    const float* wrow = W_q + (size_t)tid * E;
#pragma unroll 4
    for (int c = 0; c < E; ++c) s += att2_lds[c] * wrow[c];
    qp_lds[tid] = s;
  }
  __syncthreads();
  // fold the kp GEMM:  r = (qp @ W_k) / sqrt(E)
  if (tid < E) {
    float s = 0.f;
#pragma unroll 4
    for (int e = 0; e < E; ++e) s += qp_lds[e] * W_k[(size_t)e * E + tid];
    r_lds[tid] = s * 0.08838834764831845f;         // 1/sqrt(128)
  }
  __syncthreads();

  // logits pass (re-reads this b's 512KB slice; L2-hot): 10*tanh(X[n].r)
  for (int n = tid; n < NN; n += 256) {
    const float4* xp = (const float4*)(Xb + (size_t)n * E);
    float s = 0.f;
#pragma unroll 8
    for (int c = 0; c < E / 4; ++c) {
      float4 x = xp[c];
      s += x.x * r_lds[4 * c] + x.y * r_lds[4 * c + 1] +
           x.z * r_lds[4 * c + 2] + x.w * r_lds[4 * c + 3];
    }
    float v = 10.0f * tanhf(s);
    out[(size_t)b * NN + n] = (Mb[n] != 0) ? -INFINITY : v;
  }
}

// ---------------------------------------------------------------------------
extern "C" void kernel_launch(void* const* d_in, const int* in_sizes, int n_in,
                              void* d_out, int out_size, void* d_ws, size_t ws_size,
                              hipStream_t stream) {
  const float* node_emb   = (const float*)d_in[0];
  const float* graph_emb  = (const float*)d_in[1];
  const float* soc        = (const float*)d_in[2];
  const float* cap        = (const float*)d_in[3];
  const float* tm         = (const float*)d_in[4];
  // d_in[5] last_node_emb: unused by the reference as well
  const int*   mask       = (const int*)d_in[6];
  const float* W_ctx      = (const float*)d_in[7];
  const float* in_proj_w  = (const float*)d_in[8];
  const float* in_proj_b  = (const float*)d_in[9];
  const float* out_proj_w = (const float*)d_in[10];
  const float* out_proj_b = (const float*)d_in[11];
  const float* W_q        = (const float*)d_in[12];
  const float* W_k        = (const float*)d_in[13];
  float* out  = (float*)d_out;
  float* qbuf = (float*)d_ws;   // [B,E] fp32 = 256 KB of scratch

  ctx_query_kernel<<<BB, 128, 0, stream>>>(graph_emb, soc, cap, tm, W_ctx,
                                           in_proj_w, in_proj_b, qbuf);
  fused_attn_pointer_kernel<<<BB, 256, 0, stream>>>(node_emb, mask, in_proj_w,
                                                    in_proj_b, out_proj_w,
                                                    out_proj_b, W_q, W_k, qbuf,
                                                    out);
}
